// GNNEncoder_15229954032026
// MI455X (gfx1250) — compile-verified, matrix operand order checked by reading, not run on hardware
//
#include <hip/hip_runtime.h>

#define DEV __device__ __forceinline__

typedef __attribute__((ext_vector_type(16))) __bf16 v16bf;
typedef __attribute__((ext_vector_type(8)))  __bf16 v8bf;
typedef __attribute__((ext_vector_type(8)))  float  v8f;

union V16U { v16bf v; v8bf h[2]; };

DEV unsigned short f32_to_bf16(float f) {
    unsigned u = __float_as_uint(f);
    u = u + 0x7FFFu + ((u >> 16) & 1u);   // round-to-nearest-even
    return (unsigned short)(u >> 16);
}

DEV void atomicMaxF(float* addr, float val) {
    if (val >= 0.0f) atomicMax((int*)addr, __float_as_int(val));
    else             atomicMin((unsigned int*)addr, __float_as_uint(val));
}

// ---------------------------------------------------------------------------
// fill
__global__ void k_fill(float* __restrict__ p, float v, size_t n) {
    for (size_t i = (size_t)blockIdx.x * blockDim.x + threadIdx.x; i < n;
         i += (size_t)gridDim.x * blockDim.x) p[i] = v;
}

// convert W[K x Nn] f32 -> Wt[Nn x K] bf16 (transposed, K-contiguous rows)
__global__ void k_conv_tr(const float* __restrict__ W, unsigned short* __restrict__ Wt,
                          int K, int Nn) {
    int total = K * Nn;
    for (int i = blockIdx.x * blockDim.x + threadIdx.x; i < total;
         i += gridDim.x * blockDim.x) {
        int kk = i / Nn, nn = i % Nn;
        Wt[(size_t)nn * K + kk] = f32_to_bf16(W[i]);
    }
}

// h0 = x @ in_W + in_b + emb[node_ids]   (N x 128, bf16 mirror only)
__global__ __launch_bounds__(128) void k_embed(
    const float* __restrict__ x, const int* __restrict__ ids,
    const float* __restrict__ emb, const float* __restrict__ W,
    const float* __restrict__ bvec, unsigned short* __restrict__ hbf) {
    int n = blockIdx.x, t = threadIdx.x;
    __shared__ float xr[16];
    if (t < 16) xr[t] = x[n * 16 + t];
    __syncthreads();
    float acc = bvec[t] + emb[(size_t)ids[n] * 128 + t];
#pragma unroll
    for (int k = 0; k < 16; ++k) acc += xr[k] * W[k * 128 + t];
    hbf[(size_t)n * 128 + t] = f32_to_bf16(acc);
}

// ---------------------------------------------------------------------------
// WMMA GEMM: C[M x Nc] = A_bf16[M x K] @ Bt_bf16[Nc x K]^T (+bias)(+addend)(relu)
// One wave computes a 16x64 C strip: 1 A fragment reused across 4 B fragments
// (4 accumulators) -> 2.5 b128 loads per v_wmma_f32_16x16x32_bf16.
// Requires: M % 16 == 0, K % 32 == 0, Nc % 64 == 0.
__global__ __launch_bounds__(256) void k_wmma_gemm(
    const unsigned short* __restrict__ A, const unsigned short* __restrict__ Bt,
    const float* __restrict__ bias, const float* __restrict__ addend,
    float* __restrict__ C, unsigned short* __restrict__ Cbf,
    int M, int K, int Nc, int relu) {
    int wave = (blockIdx.x << 3) + (threadIdx.x >> 5);
    int tilesN4 = Nc >> 6;                       // groups of 4 N-tiles
    int total = (M >> 4) * tilesN4;
    if (wave >= total) return;                   // wave-uniform exit (EXEC all-1 for WMMA)
    int tm = wave / tilesN4, tg = wave % tilesN4;
    int lane = threadIdx.x & 31;
    int r = lane & 15, hi = lane >> 4;

    // A layout (16-bit 16x32): lane r (hi=0): K 0..7 / 16..23 ; hi=1: K 8..15 / 24..31
    const unsigned short* pa = A  + (size_t)(tm * 16 + r) * K + hi * 8;
    const unsigned short* pb = Bt + (size_t)(tg * 64 + r) * K + hi * 8;
    const size_t bs = (size_t)16 * K;            // stride between N-tiles of Bt

    v8f acc[4] = {};
    for (int k0 = 0; k0 < K; k0 += 32) {
        V16U a, b[4];
        a.h[0] = *reinterpret_cast<const v8bf*>(pa);
        a.h[1] = *reinterpret_cast<const v8bf*>(pa + 16);
#pragma unroll
        for (int j = 0; j < 4; ++j) {
            b[j].h[0] = *reinterpret_cast<const v8bf*>(pb + j * bs);
            b[j].h[1] = *reinterpret_cast<const v8bf*>(pb + j * bs + 16);
        }
        if (k0 + 64 < K) {
            __builtin_prefetch(pa + 64, 0, 3);
#pragma unroll
            for (int j = 0; j < 4; ++j) __builtin_prefetch(pb + j * bs + 64, 0, 3);
        }
#pragma unroll
        for (int j = 0; j < 4; ++j)
            acc[j] = __builtin_amdgcn_wmma_f32_16x16x32_bf16(
                false, a.v, false, b[j].v, (short)0, acc[j], false, false);
        pa += 32; pb += 32;
    }

    // C layout: lane<16 -> rows tm*16+0..7, lane>=16 -> rows +8..15; col = tile_n*16 + r
    int mbase = tm * 16 + hi * 8;
#pragma unroll
    for (int j = 0; j < 4; ++j) {
        int n = tg * 64 + j * 16 + r;
        float bn = bias ? bias[n] : 0.0f;
#pragma unroll
        for (int i = 0; i < 8; ++i) {
            size_t idx = (size_t)(mbase + i) * Nc + n;
            float v = acc[j][i] + bn;
            if (addend) v += addend[idx];
            if (relu) v = v > 0.0f ? v : 0.0f;
            C[idx] = v;
            if (Cbf) Cbf[idx] = f32_to_bf16(v);
        }
    }
}

// ---------------------------------------------------------------------------
// per-node attention coefficients: as[n,h] = sum_c xl[n,h,c]*att_src[h,c] (C==128)
__global__ __launch_bounds__(128) void k_att_coef(
    const float* __restrict__ xl, const float* __restrict__ att_s,
    const float* __restrict__ att_d, float* __restrict__ as_,
    float* __restrict__ ad_, int H, int C) {
    int n = blockIdx.x, t = threadIdx.x;
    __shared__ float red[128];
    for (int hd = 0; hd < H; ++hd) {
        float xv = xl[(size_t)n * H * C + hd * C + t];
        red[t] = xv * att_s[hd * C + t]; __syncthreads();
        for (int o = 64; o > 0; o >>= 1) { if (t < o) red[t] += red[t + o]; __syncthreads(); }
        if (t == 0) as_[n * H + hd] = red[0];
        __syncthreads();
        red[t] = xv * att_d[hd * C + t]; __syncthreads();
        for (int o = 64; o > 0; o >>= 1) { if (t < o) red[t] += red[t + o]; __syncthreads(); }
        if (t == 0) ad_[n * H + hd] = red[0];
        __syncthreads();
    }
}

DEV void edge_ids(const int* ei, int e, int E, int& src, int& dst) {
    if (e < E) { src = ei[e]; dst = ei[E + e]; }
    else       { src = e - E; dst = e - E; }    // self-loops
}

__global__ void k_edge_max(const int* __restrict__ ei, int E, int Nn,
                           const float* __restrict__ as_, const float* __restrict__ ad_,
                           float* __restrict__ mx, int H) {
    int total = E + Nn;
    for (int e = blockIdx.x * blockDim.x + threadIdx.x; e < total;
         e += gridDim.x * blockDim.x) {
        int src, dst; edge_ids(ei, e, E, src, dst);
        for (int h = 0; h < H; ++h) {
            float v = as_[src * H + h] + ad_[dst * H + h];
            v = v > 0.0f ? v : 0.2f * v;
            atomicMaxF(&mx[dst * H + h], v);
        }
    }
}

__global__ void k_edge_expsum(const int* __restrict__ ei, int E, int Nn,
                              const float* __restrict__ as_, const float* __restrict__ ad_,
                              const float* __restrict__ mx, float* __restrict__ sm, int H) {
    int total = E + Nn;
    for (int e = blockIdx.x * blockDim.x + threadIdx.x; e < total;
         e += gridDim.x * blockDim.x) {
        int src, dst; edge_ids(ei, e, E, src, dst);
        for (int h = 0; h < H; ++h) {
            float v = as_[src * H + h] + ad_[dst * H + h];
            v = v > 0.0f ? v : 0.2f * v;
            atomicAdd(&sm[dst * H + h], __expf(v - mx[dst * H + h]));
        }
    }
}

// y[dst, :] += xl[src, :] * alpha   (one block per edge)
__global__ __launch_bounds__(128) void k_edge_agg(
    const int* __restrict__ ei, int E, int Nn,
    const float* __restrict__ as_, const float* __restrict__ ad_,
    const float* __restrict__ mx, const float* __restrict__ sm,
    const float* __restrict__ xl, float* __restrict__ y, int H, int C) {
    int e = blockIdx.x;
    int src, dst; edge_ids(ei, e, E, src, dst);
    __shared__ float alpha[4];
    int t = threadIdx.x;
    if (t < H) {
        float v = as_[src * H + t] + ad_[dst * H + t];
        v = v > 0.0f ? v : 0.2f * v;
        alpha[t] = __expf(v - mx[dst * H + t]) / sm[dst * H + t];
    }
    __syncthreads();
    int OD = H * C;
    for (int c = t; c < OD; c += 128) {
        atomicAdd(&y[(size_t)dst * OD + c], xl[(size_t)src * OD + c] * alpha[c / C]);
    }
}

// y = relu(BN(y + bias))
__global__ void k_bias_bn_relu(float* __restrict__ y, const float* __restrict__ bias,
                               const float* __restrict__ g, const float* __restrict__ b,
                               const float* __restrict__ m, const float* __restrict__ v,
                               int Nn, int OD) {
    size_t total = (size_t)Nn * OD;
    for (size_t i = (size_t)blockIdx.x * blockDim.x + threadIdx.x; i < total;
         i += (size_t)gridDim.x * blockDim.x) {
        int c = (int)(i % OD);
        float val = y[i] + bias[c];
        val = (val - m[c]) * rsqrtf(v[c] + 1e-5f) * g[c] + b[c];
        y[i] = val > 0.0f ? val : 0.0f;
    }
}

// ---------------------------------------------------------------------------
// pooling over batch (OD == 128)
__global__ __launch_bounds__(128) void k_pool(
    const float* __restrict__ h, const int* __restrict__ batch,
    float* __restrict__ psum, float* __restrict__ pmax, float* __restrict__ cnt) {
    int n = blockIdx.x, t = threadIdx.x;
    int b = batch[n];
    float v = h[(size_t)n * 128 + t];
    atomicAdd(&psum[b * 128 + t], v);
    atomicMaxF(&pmax[b * 128 + t], v);
    if (t == 0) atomicAdd(&cnt[b], 1.0f);
}

__global__ __launch_bounds__(128) void k_pool_final(
    const float* __restrict__ psum, const float* __restrict__ pmax,
    const float* __restrict__ cnt, unsigned short* __restrict__ gbf) {
    int gi = blockIdx.x, t = threadIdx.x;
    float c = cnt[gi];
    float mean = psum[gi * 128 + t] / fmaxf(c, 1.0f);
    float mxv  = c > 0.0f ? pmax[gi * 128 + t] : 0.0f;
    gbf[gi * 256 + t]       = f32_to_bf16(mean);
    gbf[gi * 256 + 128 + t] = f32_to_bf16(mxv);
}

__global__ __launch_bounds__(256) void k_layernorm(
    const float* __restrict__ z, const float* __restrict__ g,
    const float* __restrict__ b, float* __restrict__ out, int D) {
    int rr = blockIdx.x, t = threadIdx.x;
    __shared__ float red[256];
    float s = 0.0f;
    for (int c = t; c < D; c += 256) s += z[rr * D + c];
    red[t] = s; __syncthreads();
    for (int o = 128; o > 0; o >>= 1) { if (t < o) red[t] += red[t + o]; __syncthreads(); }
    float mu = red[0] / D; __syncthreads();
    float s2 = 0.0f;
    for (int c = t; c < D; c += 256) { float d = z[rr * D + c] - mu; s2 += d * d; }
    red[t] = s2; __syncthreads();
    for (int o = 128; o > 0; o >>= 1) { if (t < o) red[t] += red[t + o]; __syncthreads(); }
    float inv = rsqrtf(red[0] / D + 1e-5f); __syncthreads();
    for (int c = t; c < D; c += 256)
        out[rr * D + c] = (z[rr * D + c] - mu) * inv * g[c] + b[c];
}

// ---------------------------------------------------------------------------
extern "C" void kernel_launch(void* const* d_in, const int* in_sizes, int n_in,
                              void* d_out, int out_size, void* d_ws, size_t ws_size,
                              hipStream_t stream) {
    (void)in_sizes; (void)n_in; (void)out_size; (void)ws_size;
    const int N = 50000, E = 800000, G = 64, OUT = 768;
    const int EE = E + N;

    const float* x     = (const float*)d_in[0];
    const int*   ids   = (const int*)d_in[1];
    const int*   ei    = (const int*)d_in[2];
    const int*   batch = (const int*)d_in[3];
    const float* emb   = (const float*)d_in[4];
    const float* in_W  = (const float*)d_in[5];
    const float* in_b  = (const float*)d_in[6];
    const float* out_W = (const float*)d_in[37];
    const float* out_b = (const float*)d_in[38];
    const float* ln_g  = (const float*)d_in[39];
    const float* ln_b  = (const float*)d_in[40];

    // workspace carve
    char* base = (char*)d_ws; size_t off = 0;
    auto carve = [&](size_t bytes) -> void* {
        void* p = base + off; off = (off + bytes + 255) & ~(size_t)255; return p;
    };
    float* xl   = (float*)carve((size_t)N * 512 * 4);   // lin out / newh f32
    float* yb   = (float*)carve((size_t)N * 512 * 4);   // attention accumulator
    unsigned short* hbfA = (unsigned short*)carve((size_t)N * 512 * 2);
    unsigned short* hbfB = (unsigned short*)carve((size_t)N * 512 * 2);
    float* as_ = (float*)carve((size_t)N * 4 * 4);
    float* ad_ = (float*)carve((size_t)N * 4 * 4);
    float* mx  = (float*)carve((size_t)N * 4 * 4);
    float* sm  = (float*)carve((size_t)N * 4 * 4);
    unsigned short* wlin = (unsigned short*)carve(512 * 512 * 2);
    unsigned short* wres = (unsigned short*)carve(512 * 512 * 2);
    unsigned short* wout = (unsigned short*)carve(768 * 256 * 2);
    float* psum = (float*)carve(64 * 128 * 4);
    float* pmax = (float*)carve(64 * 128 * 4);
    float* cnt  = (float*)carve(64 * 4);
    unsigned short* gbf = (unsigned short*)carve(64 * 256 * 2);
    float* zbuf = (float*)carve(64 * 768 * 4);

    auto gemm = [&](const unsigned short* A, const unsigned short* Bt,
                    const float* bias, const float* add, float* C,
                    unsigned short* Cb, int M, int K, int Nc, int relu) {
        int waves = (M / 16) * (Nc / 64);   // one wave per 16x64 C strip
        k_wmma_gemm<<<(waves + 7) / 8, 256, 0, stream>>>(A, Bt, bias, add, C, Cb,
                                                         M, K, Nc, relu);
    };

    // input projection + embedding -> bf16 h0
    k_embed<<<N, 128, 0, stream>>>(x, ids, emb, in_W, in_b, hbfA);

    unsigned short* hcur = hbfA;
    unsigned short* hnxt = hbfB;
    const int pdim[3] = {128, 512, 512};
    const int Hh[3]   = {4, 4, 1};
    const int ODd[3]  = {512, 512, 128};

    for (int L = 0; L < 3; ++L) {
        const float* linW = (const float*)d_in[7 + L * 10 + 0];
        const float* atS  = (const float*)d_in[7 + L * 10 + 1];
        const float* atD  = (const float*)d_in[7 + L * 10 + 2];
        const float* bia  = (const float*)d_in[7 + L * 10 + 3];
        const float* bng  = (const float*)d_in[7 + L * 10 + 4];
        const float* bnb  = (const float*)d_in[7 + L * 10 + 5];
        const float* bnm  = (const float*)d_in[7 + L * 10 + 6];
        const float* bnv  = (const float*)d_in[7 + L * 10 + 7];
        const float* resW = (const float*)d_in[7 + L * 10 + 8];
        const float* resb = (const float*)d_in[7 + L * 10 + 9];
        int P = pdim[L], H = Hh[L], C = 128, OD = ODd[L];

        k_conv_tr<<<(P * OD + 255) / 256, 256, 0, stream>>>(linW, wlin, P, OD);
        k_conv_tr<<<(P * OD + 255) / 256, 256, 0, stream>>>(resW, wres, P, OD);

        // xl = h @ lin_W
        gemm(hcur, wlin, nullptr, nullptr, xl, nullptr, N, P, OD, 0);

        k_att_coef<<<N, 128, 0, stream>>>(xl, atS, atD, as_, ad_, H, C);
        k_fill<<<256, 256, 0, stream>>>(mx, -INFINITY, (size_t)N * H);
        k_fill<<<256, 256, 0, stream>>>(sm, 0.0f, (size_t)N * H);
        k_fill<<<8192, 256, 0, stream>>>(yb, 0.0f, (size_t)N * OD);

        k_edge_max<<<(EE + 255) / 256, 256, 0, stream>>>(ei, E, N, as_, ad_, mx, H);
        k_edge_expsum<<<(EE + 255) / 256, 256, 0, stream>>>(ei, E, N, as_, ad_, mx, sm, H);
        k_edge_agg<<<EE, 128, 0, stream>>>(ei, E, N, as_, ad_, mx, sm, xl, yb, H, C);

        k_bias_bn_relu<<<8192, 256, 0, stream>>>(yb, bia, bng, bnb, bnm, bnv, N, OD);

        // newh = h @ res_W + res_b + y   (f32 into xl, bf16 mirror into hnxt)
        gemm(hcur, wres, resb, yb, xl, hnxt, N, P, OD, 0);
        unsigned short* t = hcur; hcur = hnxt; hnxt = t;
    }

    // pooling: final h (f32) lives in xl, N x 128
    k_fill<<<64, 256, 0, stream>>>(psum, 0.0f, (size_t)G * 128);
    k_fill<<<64, 256, 0, stream>>>(pmax, -INFINITY, (size_t)G * 128);
    k_fill<<<1, 64, 0, stream>>>(cnt, 0.0f, (size_t)G);
    k_pool<<<N, 128, 0, stream>>>(xl, batch, psum, pmax, cnt);
    k_pool_final<<<G, 128, 0, stream>>>(psum, pmax, cnt, gbf);

    // z = relu(g @ out_W + out_b) ; then layernorm -> d_out
    k_conv_tr<<<(256 * 768 + 255) / 256, 256, 0, stream>>>(out_W, wout, 256, 768);
    gemm(gbf, wout, out_b, nullptr, zbuf, nullptr, G, 256, 768, 1);
    k_layernorm<<<G, 256, 0, stream>>>(zbuf, ln_g, ln_b, (float*)d_out, OUT);
}
